// GraphAttentionLayer_45260365365325
// MI455X (gfx1250) — matrix operand order repaired and checked by reading, not compiled, not run on hardware
//
#include <hip/hip_runtime.h>
#include <hip/hip_fp16.h>
#include <math.h>

typedef __attribute__((ext_vector_type(16))) _Float16 v16h;
typedef __attribute__((ext_vector_type(8)))  _Float16 v8h;
typedef __attribute__((ext_vector_type(4)))  _Float16 v4h;
typedef __attribute__((ext_vector_type(8)))  float    v8f;

#define N_NODES 4096
#define F_DIM   512
#define NHEAD   8
#define HDIM    64

__device__ __forceinline__ v8f wmma_f16(v16h a, v16h b, v8f c) {
  // v_wmma_f32_16x16x32_f16 : D = A(16x32) * B(32x16) + C(16x16)
  return __builtin_amdgcn_wmma_f32_16x16x32_f16(false, a, false, b, (short)0, c,
                                                false, false);
}

// ---------------------------------------------------------------------------
// f32 -> f16 conversion (vectorized by 4)
// ---------------------------------------------------------------------------
__global__ void cvt_f32_to_f16(const float* __restrict__ src,
                               _Float16* __restrict__ dst, int n) {
  int i = (blockIdx.x * blockDim.x + threadIdx.x) * 4;
  if (i < n) {
    float4 v = *(const float4*)(src + i);
    v4h h;
    h[0] = (_Float16)v.x; h[1] = (_Float16)v.y;
    h[2] = (_Float16)v.z; h[3] = (_Float16)v.w;
    *(v4h*)(dst + i) = h;
  }
}

// ---------------------------------------------------------------------------
// Fused QKV projection: C = xh @ W^T  (nn.Linear), one wave per 16x64 tile.
// which==0 -> Qh row-major, 1 -> Kh row-major, 2 -> V stored transposed (Vt).
// ---------------------------------------------------------------------------
__global__ __launch_bounds__(32) void proj_kernel(
    const _Float16* __restrict__ xh, const _Float16* __restrict__ Wh,
    _Float16* __restrict__ Qh, _Float16* __restrict__ Kh,
    _Float16* __restrict__ Vt) {
  const int lane = threadIdx.x & 31;
  const int lr = lane & 15, hi = lane >> 4;
  const int i0 = blockIdx.x * 16;
  const int n0 = blockIdx.y * 64;
  const int which = blockIdx.z;
  const _Float16* W = Wh + (size_t)which * F_DIM * F_DIM;

  const v8f zf = {0.f,0.f,0.f,0.f,0.f,0.f,0.f,0.f};
  v8f acc[4] = {zf, zf, zf, zf};

  const _Float16* xrow = xh + (size_t)(i0 + lr) * F_DIM;
#pragma unroll 4
  for (int k0 = 0; k0 < F_DIM; k0 += 32) {
    // A fragment (16x32 f16): lane<16 rows hold K 0..15, hi lanes K 16..31
    v8h p0 = *(const v8h*)(xrow + k0 + hi * 16);
    v8h p1 = *(const v8h*)(xrow + k0 + hi * 16 + 8);
    v16h a;
#pragma unroll
    for (int e = 0; e < 8; ++e) { a[e] = p0[e]; a[8 + e] = p1[e]; }
#pragma unroll
    for (int nt = 0; nt < 4; ++nt) {
      // B fragment (32x16): column n = W row (n0+nt*16+lr), contiguous K
      const v16h* bp =
          (const v16h*)(W + (size_t)(n0 + nt * 16 + lr) * F_DIM + k0 + hi * 16);
      acc[nt] = wmma_f16(a, *bp, acc[nt]);
    }
  }

  if (which < 2) {
    _Float16* dst = (which == 0) ? Qh : Kh;
#pragma unroll
    for (int nt = 0; nt < 4; ++nt)
#pragma unroll
      for (int r = 0; r < 8; ++r)
        dst[(size_t)(i0 + r + hi * 8) * F_DIM + n0 + nt * 16 + lr] =
            (_Float16)acc[nt][r];
  } else {
    // Vt[f][n] = V[n][f] : lane holds 8 consecutive n for fixed f -> b128 store
#pragma unroll
    for (int nt = 0; nt < 4; ++nt) {
      v8h pk;
#pragma unroll
      for (int r = 0; r < 8; ++r) pk[r] = (_Float16)acc[nt][r];
      *(v8h*)(Vt + (size_t)(n0 + nt * 16 + lr) * N_NODES + i0 + hi * 8) = pk;
    }
  }
}

// ---------------------------------------------------------------------------
// Masked attention, flash-style online softmax. One wave per (16-row, head).
// Computes S^T = K.Q^T so P lands directly in the A-fragment layout of P@V.
// ---------------------------------------------------------------------------
__global__ __launch_bounds__(32) void attn_kernel(
    const _Float16* __restrict__ Qh, const _Float16* __restrict__ Kh,
    const _Float16* __restrict__ Vt, const int* __restrict__ adj,
    const float* __restrict__ x, float* __restrict__ out) {
  const int lane = threadIdx.x & 31;
  const int lr = lane & 15, hi = lane >> 4;
  const int i0 = blockIdx.x * 16;
  const int h  = blockIdx.y;
  const float scale = 0.04419417382415922f;  // 1/sqrt(512)

  // Resident Q B-fragments: B[k][n] = Q[i0+n][h*64+k], contiguous per lane
  const _Float16* qrow = Qh + (size_t)(i0 + lr) * F_DIM + h * HDIM;
  const v16h bq0 = *(const v16h*)(qrow + hi * 16);
  const v16h bq1 = *(const v16h*)(qrow + 32 + hi * 16);

  const v8f zf = {0.f,0.f,0.f,0.f,0.f,0.f,0.f,0.f};
  v8f o[4] = {zf, zf, zf, zf};
  float mrow = -__builtin_inff();
  float lrow = 0.0f;

  const int* adjrow = adj + ((size_t)h * N_NODES + (i0 + lr)) * N_NODES;

  for (int j0 = 0; j0 < N_NODES; j0 += 32) {
    // ---- S^T tiles: rows = j, cols = i. Lane holds row i = lr, 8 j's/reg.
    v8f s[2];
#pragma unroll
    for (int t = 0; t < 2; ++t) {
      const _Float16* krow = Kh + (size_t)(j0 + 16 * t + lr) * F_DIM + h * HDIM;
      v8h p0 = *(const v8h*)(krow + hi * 16);
      v8h p1 = *(const v8h*)(krow + hi * 16 + 8);
      v8h p2 = *(const v8h*)(krow + 32 + hi * 16);
      v8h p3 = *(const v8h*)(krow + 40 + hi * 16);
      v16h a0, a1;
#pragma unroll
      for (int e = 0; e < 8; ++e) {
        a0[e] = p0[e]; a0[8 + e] = p1[e];
        a1[e] = p2[e]; a1[8 + e] = p3[e];
      }
      v8f c = zf;
      c = wmma_f16(a0, bq0, c);
      c = wmma_f16(a1, bq1, c);
      s[t] = c;
    }

    // ---- mask + scale. Lane's j for tile t: j0 + 16t + hi*8 + r (contig.)
    float sv[16];
    float smax = -__builtin_inff();
#pragma unroll
    for (int t = 0; t < 2; ++t) {
      const int4* ap = (const int4*)(adjrow + j0 + 16 * t + hi * 8);
      int4 m0 = ap[0], m1 = ap[1];
      int mm[8] = {m0.x, m0.y, m0.z, m0.w, m1.x, m1.y, m1.z, m1.w};
#pragma unroll
      for (int r = 0; r < 8; ++r) {
        float v = (mm[r] > 0) ? s[t][r] * scale : -9.0e15f;
        sv[t * 8 + r] = v;
        smax = fmaxf(smax, v);
      }
    }
    if (j0 + 32 < N_NODES)
      __builtin_prefetch(adjrow + j0 + 32 + hi * 8, 0, 1);

    // ---- online softmax update (row i lives in lanes lr and lr+16)
    smax = fmaxf(smax, __shfl_xor(smax, 16, 32));
    float mnew = fmaxf(mrow, smax);
    float corr = __expf(mrow - mnew);
    float psum = 0.f;
    v16h pa;  // exp'd tile IS the A-fragment for P@V (16x32 f16)
#pragma unroll
    for (int e = 0; e < 16; ++e) {
      float p = __expf(sv[e] - mnew);
      psum += p;
      pa[e] = (_Float16)p;
    }
    psum += __shfl_xor(psum, 16, 32);
    lrow = lrow * corr + psum;
    mrow = mnew;

    // ---- rescale O rows (O C-frag row M = r + hi*8; corr sits in lane M)
    float cb[8];
#pragma unroll
    for (int r = 0; r < 8; ++r) cb[r] = __shfl(corr, r + hi * 8, 32);
#pragma unroll
    for (int nt = 0; nt < 4; ++nt)
#pragma unroll
      for (int r = 0; r < 8; ++r) o[nt][r] *= cb[r];

    // ---- O += P @ V : B fragment = Vt row (feature), j = hi*8+0..7 & 16+hi*8+0..7
#pragma unroll
    for (int nt = 0; nt < 4; ++nt) {
      const _Float16* vrow = Vt + (size_t)(h * HDIM + nt * 16 + lr) * N_NODES + j0;
      v8h q0 = *(const v8h*)(vrow + hi * 8);
      v8h q1 = *(const v8h*)(vrow + 16 + hi * 8);
      v16h bv;
#pragma unroll
      for (int e = 0; e < 8; ++e) { bv[e] = q0[e]; bv[8 + e] = q1[e]; }
      o[nt] = wmma_f16(pa, bv, o[nt]);
    }
  }

  // ---- epilogue: O/l + residual, ELU
  float linv = 1.0f / lrow;
  float lb[8];
#pragma unroll
  for (int r = 0; r < 8; ++r) lb[r] = __shfl(linv, r + hi * 8, 32);
#pragma unroll
  for (int nt = 0; nt < 4; ++nt) {
#pragma unroll
    for (int r = 0; r < 8; ++r) {
      int i = i0 + r + hi * 8;
      int f = h * HDIM + nt * 16 + lr;
      size_t idx = (size_t)i * F_DIM + f;
      float v = o[nt][r] * lb[r] + x[idx];
      out[idx] = (v > 0.f) ? v : expm1f(v);
    }
  }
}

// ---------------------------------------------------------------------------
extern "C" void kernel_launch(void* const* d_in, const int* in_sizes, int n_in,
                              void* d_out, int out_size, void* d_ws,
                              size_t ws_size, hipStream_t stream) {
  const float* x  = (const float*)d_in[0];
  const float* Wq = (const float*)d_in[1];
  const float* Wk = (const float*)d_in[2];
  const float* Wv = (const float*)d_in[3];
  const int*  adj = (const int*)d_in[4];
  float* out = (float*)d_out;

  char* ws = (char*)d_ws;
  _Float16* xh = (_Float16*)ws;  ws += (size_t)N_NODES * F_DIM * sizeof(_Float16);
  _Float16* Wh = (_Float16*)ws;  ws += (size_t)3 * F_DIM * F_DIM * sizeof(_Float16);
  _Float16* Qh = (_Float16*)ws;  ws += (size_t)N_NODES * F_DIM * sizeof(_Float16);
  _Float16* Kh = (_Float16*)ws;  ws += (size_t)N_NODES * F_DIM * sizeof(_Float16);
  _Float16* Vt = (_Float16*)ws;  ws += (size_t)N_NODES * F_DIM * sizeof(_Float16);
  (void)ws_size; (void)in_sizes; (void)n_in; (void)out_size;

  // 1) f32 -> f16 conversions
  {
    int n = N_NODES * F_DIM;                    // 2,097,152
    cvt_f32_to_f16<<<(n / 4 + 255) / 256, 256, 0, stream>>>(x, xh, n);
    int m = F_DIM * F_DIM;                      // 262,144
    cvt_f32_to_f16<<<(m / 4 + 255) / 256, 256, 0, stream>>>(Wq, Wh, m);
    cvt_f32_to_f16<<<(m / 4 + 255) / 256, 256, 0, stream>>>(Wk, Wh + m, m);
    cvt_f32_to_f16<<<(m / 4 + 255) / 256, 256, 0, stream>>>(Wv, Wh + 2 * m, m);
  }
  // 2) QKV projections (one wave per 16x64 tile; z = which matrix)
  proj_kernel<<<dim3(N_NODES / 16, F_DIM / 64, 3), 32, 0, stream>>>(
      xh, Wh, Qh, Kh, Vt);
  // 3) masked flash attention + residual + ELU (one wave per 16 rows x head)
  attn_kernel<<<dim3(N_NODES / 16, NHEAD), 32, 0, stream>>>(
      Qh, Kh, Vt, adj, x, out);
}
